// SeqAttention_8839042695766
// MI455X (gfx1250) — matrix operand order repaired and checked by reading, not compiled, not run on hardware
//
#include <hip/hip_runtime.h>
#include <hip/hip_bf16.h>
#include <math.h>

#define BS     32
#define SEQ    2048
#define ENCD   512
#define IH     1024
#define HEADS  8

typedef __attribute__((ext_vector_type(16))) __bf16 v16bf;
typedef __attribute__((ext_vector_type(8)))  __bf16 v8bf;
typedef __attribute__((ext_vector_type(8)))  float  v8f;
typedef __attribute__((ext_vector_type(4)))  unsigned int u32x4;
typedef __attribute__((ext_vector_type(8)))  unsigned int u32x8;

union bf16x16 { v16bf v; v8bf h[2]; };

__device__ __forceinline__ float softplusf(float x) {
    return (x > 15.f) ? x : log1pf(__expf(x));
}

// ---------------------------------------------------------------------------
// TDM: 1-D contiguous tile (nelem f32) global -> LDS, via tensor descriptor
// (CDNA5 ISA). Wave-level op; EXEC ignored.
//   group0: count=1 | lds_addr | global_addr | type=2
//   group1: data_size=4B, tensor_dim0=nelem, tensor_dim1=1, tile_dim0=nelem,
//           tile_dim1=1, stride0=nelem
// ---------------------------------------------------------------------------
__device__ __forceinline__ void tdm_load_1d_f32(unsigned int lds_off,
                                                const float* gptr,
                                                unsigned int nelem) {
    unsigned long long ga = (unsigned long long)(size_t)gptr;
    u32x4 g0;
    g0[0] = 1u;                                    // count=1, is_restore=0
    g0[1] = lds_off;                               // lds_addr (bytes)
    g0[2] = (unsigned int)ga;                      // global_addr[31:0]
    g0[3] = (unsigned int)(ga >> 32) | (2u << 30); // global_addr[56:32] | type=2
    u32x8 g1;
    g1[0] = (2u << 16);                            // data_size=4B, no flags
    g1[1] = (nelem & 0xFFFFu) << 16;               // tensor_dim0[15:0]
    g1[2] = (nelem >> 16) | (1u << 16);            // tensor_dim0[31:16] | tensor_dim1=1
    g1[3] = (nelem & 0xFFFFu) << 16;               // tensor_dim1 hi=0 | tile_dim0
    g1[4] = 1u;                                    // tile_dim1=1, tile_dim2=0
    g1[5] = nelem;                                 // tensor_dim0_stride[31:0]
    g1[6] = 0u;
    g1[7] = 0u;
    asm volatile("tensor_load_to_lds %0, %1" :: "s"(g0), "s"(g1) : "memory");
}

// ---------------------------------------------------------------------------
// Kernel 1: per-batch prep. query = dec@Wq + bq + bv (bv folded). beta/kappa.
// ---------------------------------------------------------------------------
__global__ __launch_bounds__(256) void prep_kernel(
    const float* __restrict__ dec, const float* __restrict__ kappa_in,
    const float* __restrict__ Wq, const float* __restrict__ bq,
    const float* __restrict__ bv,
    const float* __restrict__ Wb, const float* __restrict__ bb,
    const float* __restrict__ Wk, const float* __restrict__ bk,
    float* __restrict__ query, float* __restrict__ beta,
    float* __restrict__ kappa_ws, float* __restrict__ kappa_out)
{
    __shared__ float ds_[IH];
    const int b = blockIdx.x, t = threadIdx.x;
    for (int j = 0; j < 4; ++j) ds_[t + j*256] = dec[b*IH + t + j*256];
    __syncthreads();
    for (int j = 0; j < 4; ++j) {
        int n = t + j*256;
        float acc = 0.f;
        for (int k = 0; k < IH; ++k) acc = fmaf(ds_[k], Wq[k*IH + n], acc);
        query[b*IH + n] = acc + bq[n] + bv[n];
    }
    if (t < HEADS) {
        float acc = 0.f;
        for (int k = 0; k < IH; ++k) acc = fmaf(ds_[k], Wb[k*HEADS + t], acc);
        beta[b*HEADS + t] = softplusf(acc + bb[t]);
    } else if (t < 2*HEADS) {
        int h = t - HEADS;
        float acc = 0.f;
        for (int k = 0; k < IH; ++k) acc = fmaf(ds_[k], Wk[k*HEADS + h], acc);
        float kp = kappa_in[b*HEADS + h] + softplusf(acc + bk[h]);
        kappa_ws[b*HEADS + h]  = kp;
        kappa_out[b*HEADS + h] = kp;   // directly into d_out kappa region
    }
}

// ---------------------------------------------------------------------------
// Kernel 2: bf16 operand prep. WvT[n][k] (1024x512); WfT[h][n] (16x1024,
// heads padded with zeros) so WMMA A/B fragments are contiguous-K loads.
// ---------------------------------------------------------------------------
__global__ __launch_bounds__(256) void convert_kernel(
    const float* __restrict__ Wv, const float* __restrict__ Wf,
    __bf16* __restrict__ WvT, __bf16* __restrict__ WfT)
{
    const int t = threadIdx.x;
    if (blockIdx.x < 1024) {
        int n = blockIdx.x;
        for (int j = 0; j < 2; ++j) {
            int k = t + j*256;
            WvT[n*ENCD + k] = (__bf16)Wv[k*IH + n];
        }
    } else {
        for (int j = 0; j < 64; ++j) {
            int idx = t + j*256;
            int h = idx >> 10, n = idx & 1023;
            WfT[idx] = (h < HEADS) ? (__bf16)Wf[n*HEADS + h] : (__bf16)0.f;
        }
    }
}

// ---------------------------------------------------------------------------
// Kernel 3 (main, fused): 64-row enc tile per block (halves L2 Wv traffic
// vs 32 rows). value = enc@Wv (WMMA bf16/f32acc) -> +query -> tanh ->
// alpha = tanh_mat@Wf (second WMMA) -> softplus*mask*Gaussian -> score.
// 8 waves = 4 row-groups x 2 n-halves of the 1024 value columns.
// LDS ~79KB (gfx1250: up to 320KB per workgroup).
// ---------------------------------------------------------------------------
__global__ __launch_bounds__(256) void fused_kernel(
    const float* __restrict__ enc, const float* __restrict__ mask,
    const __bf16* __restrict__ WvT, const __bf16* __restrict__ WfT,
    const float* __restrict__ query, const float* __restrict__ beta,
    const float* __restrict__ kappa, const float* __restrict__ bf_,
    float* __restrict__ score_out)
{
    __shared__ __bf16 encs[64][520];      // 66,560 B, padded rows (bank spread)
    __shared__ __bf16 stage[8][16][40];   // 10,240 B wave-private tanh restage
    __shared__ float  alpha_s[64][HEADS]; //  2,048 B

    const int b  = blockIdx.x >> 5;
    const int s0 = (blockIdx.x & 31) * 64;
    const int t  = threadIdx.x;

    // --- cooperative: enc tile f32 -> bf16 LDS; zero alpha accumulators ---
    const float* encb = enc + (size_t)(b*SEQ + s0) * ENCD;
    for (int j = 0; j < 32; ++j) {
        int idx = (j*256 + t) * 4;
        int r = idx >> 9, c = idx & 511;
        float4 v = *(const float4*)(encb + idx);
        encs[r][c+0] = (__bf16)v.x;  encs[r][c+1] = (__bf16)v.y;
        encs[r][c+2] = (__bf16)v.z;  encs[r][c+3] = (__bf16)v.w;
    }
    for (int j = 0; j < 2; ++j) {
        int idx = t + j*256;
        alpha_s[idx >> 3][idx & 7] = 0.f;
    }
    __syncthreads();

    // --- per-wave WMMA pipeline ---
    const int w    = t >> 5;
    const int l    = t & 31;
    const int lh   = l >> 4;       // lane half: selects 16-wide K half (contiguous K order)
    const int ll   = l & 15;
    const int rg   = w >> 1;       // 4 row groups
    const int row0 = rg * 16;
    const int qoff = (w & 1) * 512;
    const int am   = row0 + ll;
    const float* qrow = query + b*IH;

    v8f aacc = {};
    for (int it = 0; it < 16; ++it) {
        const int nb0 = qoff + it*32;
        v8f acc0 = {}, acc1 = {};
        const __bf16* w0 = WvT + (size_t)(nb0 + ll)      * ENCD;
        const __bf16* w1 = WvT + (size_t)(nb0 + 16 + ll) * ENCD;
        #pragma unroll
        for (int k = 0; k < ENCD; k += 32) {
            bf16x16 a;
            a.h[0] = *(const v8bf*)&encs[am][k + lh*16];      // A: K {0-7 | 16-23}
            a.h[1] = *(const v8bf*)&encs[am][k + lh*16 + 8];  // A: K {8-15 | 24-31}
            int ko = k + lh*16;                                // B: K {0-15 | 16-31}
            v16bf b0 = *(const v16bf*)(w0 + ko);
            v16bf b1 = *(const v16bf*)(w1 + ko);
            acc0 = __builtin_amdgcn_wmma_f32_16x16x32_bf16(false, a.v, false, b0,
                                                           (short)0, acc0, false, false);
            acc1 = __builtin_amdgcn_wmma_f32_16x16x32_bf16(false, a.v, false, b1,
                                                           (short)0, acc1, false, false);
        }
        // epilogue: +query, tanh, restage as bf16 A-matrix for head reduction
        float q0 = qrow[nb0 + ll];
        float q1 = qrow[nb0 + 16 + ll];
        #pragma unroll
        for (int i = 0; i < 8; ++i) {
            int m = i + lh*8;                      // C layout: M = i + 8*(lane/16)
            stage[w][m][ll]      = (__bf16)tanhf(acc0[i] + q0);
            stage[w][m][16 + ll] = (__bf16)tanhf(acc1[i] + q1);
        }
        asm volatile("s_wait_dscnt 0" ::: "memory");  // cross-lane LDS restage fence
        bf16x16 aa;
        aa.h[0] = *(const v8bf*)&stage[w][ll][lh*16];
        aa.h[1] = *(const v8bf*)&stage[w][ll][lh*16 + 8];
        v16bf fb = *(const v16bf*)(WfT + ll*IH + nb0 + lh*16);
        aacc = __builtin_amdgcn_wmma_f32_16x16x32_bf16(false, aa.v, false, fb,
                                                       (short)0, aacc, false, false);
    }

    // combine the 2 n-half waves per row group (heads 8..15 are exact zeros)
    #pragma unroll
    for (int i = 0; i < 8; ++i) {
        if (ll < HEADS) atomicAdd(&alpha_s[row0 + i + lh*8][ll], aacc[i]);
    }
    __syncthreads();

    // --- score: two (row, head) entries per thread ---
    for (int j = 0; j < 2; ++j) {
        int idx = t + j*256;
        int row = idx >> 3, h = idx & 7;
        float x = alpha_s[row][h] + bf_[h];
        float a = softplusf(x) * mask[b*SEQ + s0 + row];
        float kp = kappa[b*HEADS + h];
        float be = beta[b*HEADS + h];
        float d  = kp - (float)(s0 + row);
        score_out[(size_t)(b*SEQ + s0 + row)*HEADS + h] = a * __expf(-be * d * d);
    }
}

// ---------------------------------------------------------------------------
// Kernel 4: deterministic context einsum  ctx[b][h][e] = sum_s score*enc.
// Score chunks (8KB, contiguous) staged into LDS by the Tensor Data Mover,
// double buffered: wave 0 issues tensor_load_to_lds for chunk c+1, waits
// TENSORcnt<=1 for chunk c, then the block streams enc with VALU FMAs.
// ---------------------------------------------------------------------------
__global__ __launch_bounds__(256) void ctx_kernel(
    const float* __restrict__ enc, const float* __restrict__ score,
    float* __restrict__ ctx)
{
    __shared__ float ss[2][256 * HEADS];   // 2 x 8KB score chunks
    const int b = blockIdx.x >> 2, ec = blockIdx.x & 3, t = threadIdx.x;
    const int w  = t >> 5;
    const int e  = ec*128 + (t & 127);
    const int hb = (t >> 7) * 4;
    const float* sbase = score + (size_t)b * SEQ * HEADS;

    if (w == 0) {
        tdm_load_1d_f32((unsigned int)(size_t)&ss[0][0], sbase, 256*HEADS);
    }
    float a0 = 0.f, a1 = 0.f, a2 = 0.f, a3 = 0.f;
    for (int c = 0; c < 8; ++c) {
        if (w == 0) {
            if (c < 7) {
                tdm_load_1d_f32((unsigned int)(size_t)&ss[(c+1)&1][0],
                                sbase + (size_t)(c+1)*256*HEADS, 256*HEADS);
                __builtin_amdgcn_s_wait_tensorcnt(1);   // chunk c complete
            } else {
                __builtin_amdgcn_s_wait_tensorcnt(0);
            }
        }
        __syncthreads();                                 // chunk c visible to all
        const float* sc = ss[c & 1];
        const int sb = c * 256;
        for (int s = 0; s < 256; ++s) {
            float ev = enc[(size_t)(b*SEQ + sb + s)*ENCD + e];
            a0 = fmaf(sc[s*HEADS + hb + 0], ev, a0);
            a1 = fmaf(sc[s*HEADS + hb + 1], ev, a1);
            a2 = fmaf(sc[s*HEADS + hb + 2], ev, a2);
            a3 = fmaf(sc[s*HEADS + hb + 3], ev, a3);
        }
        __syncthreads();                                 // done reading before reuse
    }
    ctx[b*4096 + (hb+0)*ENCD + e] = a0;
    ctx[b*4096 + (hb+1)*ENCD + e] = a1;
    ctx[b*4096 + (hb+2)*ENCD + e] = a2;
    ctx[b*4096 + (hb+3)*ENCD + e] = a3;
}

// ---------------------------------------------------------------------------
// Kernel 5: out[b] = ctx[b] @ Wfc + bfc   (32x4096 @ 4096x512)
// ---------------------------------------------------------------------------
__global__ __launch_bounds__(256) void out_kernel(
    const float* __restrict__ ctx, const float* __restrict__ Wfc,
    const float* __restrict__ bfc, float* __restrict__ out)
{
    __shared__ float cs[4096];
    const int b = blockIdx.x, t = threadIdx.x;
    for (int j = 0; j < 16; ++j) cs[t + j*256] = ctx[b*4096 + t + j*256];
    __syncthreads();
    for (int j = 0; j < 2; ++j) {
        int e = t + j*256;
        float acc = bfc[e];
        for (int k = 0; k < 4096; ++k) acc = fmaf(cs[k], Wfc[k*ENCD + e], acc);
        out[b*ENCD + e] = acc;
    }
}

// ---------------------------------------------------------------------------
extern "C" void kernel_launch(void* const* d_in, const int* in_sizes, int n_in,
                              void* d_out, int out_size, void* d_ws, size_t ws_size,
                              hipStream_t stream) {
    const float* enc   = (const float*)d_in[0];
    const float* dec   = (const float*)d_in[1];
    const float* kappa = (const float*)d_in[2];
    const float* mask  = (const float*)d_in[3];
    const float* Wv    = (const float*)d_in[4];
    const float* bv    = (const float*)d_in[5];
    const float* Wq    = (const float*)d_in[6];
    const float* bq    = (const float*)d_in[7];
    const float* Wf    = (const float*)d_in[8];
    const float* bf_   = (const float*)d_in[9];
    const float* Wb    = (const float*)d_in[10];
    const float* bb    = (const float*)d_in[11];
    const float* Wk    = (const float*)d_in[12];
    const float* bk    = (const float*)d_in[13];
    const float* Wfc   = (const float*)d_in[14];
    const float* bfc   = (const float*)d_in[15];

    float* out = (float*)d_out;
    float* outCtx   = out;                 // (32,1,512)
    float* outKappa = out + 16384;         // (32,1,8)
    float* outScore = out + 16384 + 256;   // (32,2048,8)

    char* ws = (char*)d_ws;
    float*  query = (float*)(ws + 0);          // 32*1024 f32
    float*  beta  = (float*)(ws + 131072);     // 32*8 f32
    float*  kapws = (float*)(ws + 132096);     // 32*8 f32
    float*  ctx   = (float*)(ws + 133120);     // 32*4096 f32
    __bf16* WvT   = (__bf16*)(ws + 657408);    // 1024*512 bf16
    __bf16* WfT   = (__bf16*)(ws + 1705984);   // 16*1024 bf16

    prep_kernel<<<BS, 256, 0, stream>>>(dec, kappa, Wq, bq, bv, Wb, bb, Wk, bk,
                                        query, beta, kapws, outKappa);
    convert_kernel<<<1025, 256, 0, stream>>>(Wv, Wf, WvT, WfT);
    fused_kernel<<<BS * (SEQ/64), 256, 0, stream>>>(enc, mask, WvT, WfT, query,
                                                    beta, kapws, bf_, outScore);
    ctx_kernel<<<BS * 4, 256, 0, stream>>>(enc, outScore, ctx);
    out_kernel<<<BS, 256, 0, stream>>>(ctx, Wfc, bfc, outCtx);
}
